// MGM_28862180229693
// MI455X (gfx1250) — matrix operand order repaired and checked by reading, not compiled, not run on hardware
//
#include <hip/hip_runtime.h>
#include <hip/hip_bf16.h>
#include <math.h>

typedef __attribute__((ext_vector_type(16))) _Float16 v16h;
typedef __attribute__((ext_vector_type(8)))  _Float16 v8h;
typedef __attribute__((ext_vector_type(4)))  _Float16 v4h;
typedef __attribute__((ext_vector_type(8)))  float    v8f;
typedef __attribute__((ext_vector_type(4)))  float    v4f;

#define B_  32
#define T_  512
#define D_  512
#define H_  512
#define M_TOT (B_ * T_)                          // 16384 rows of the big GEMM
static const size_t PLANE = (size_t)M_TOT * H_;  // 8,388,608 elements per f32 plane

// LDS staging geometry for the GEMM B panel
#define CHUNK_K   128                 // K elements staged per buffer
#define BPAD      8                   // pad halves -> row stride 136 (bank-conflict free)
#define BSTRIDE   (CHUNK_K + BPAD)    // 136 halves per column row
#define NSTEPS    (D_ / 32)           // 16 K-steps of 32
#define NCHUNK    (D_ / CHUNK_K)      // 4 chunks

// ---------------------------------------------------------------------------
// Fragment loaders
// ---------------------------------------------------------------------------
__device__ __forceinline__ v16h load_frag16(const _Float16* p0, const _Float16* p1)
{
    const v8h a0 = *(const v8h*)p0;
    const v8h a1 = *(const v8h*)p1;
    return __builtin_shufflevector(a0, a1,
        0, 1, 2, 3, 4, 5, 6, 7, 8, 9, 10, 11, 12, 13, 14, 15);
}

// ---------------------------------------------------------------------------
// Kernel 0a: convert x (f32) -> f16, 4 elements per thread, fully coalesced.
// ---------------------------------------------------------------------------
__global__ __launch_bounds__(256) void convert_x_kernel(
    const float* __restrict__ x, _Float16* __restrict__ xh)
{
    const size_t i = ((size_t)blockIdx.x * 256 + threadIdx.x) * 4;
    const v4f v = *(const v4f*)(x + i);
    v4h o;
    o[0] = (_Float16)v[0];
    o[1] = (_Float16)v[1];
    o[2] = (_Float16)v[2];
    o[3] = (_Float16)v[3];
    *(v4h*)(xh + i) = o;
}

// ---------------------------------------------------------------------------
// Kernel 0b: transpose + convert the three weight matrices.
// W[p] is [D_, H_] row-major f32; output wt[p] is [H_, D_] row-major f16
// so a WMMA B-fragment (one output column, 16 consecutive K) is contiguous.
// ---------------------------------------------------------------------------
__global__ __launch_bounds__(256) void transpose_w_kernel(
    const float* __restrict__ Wu, const float* __restrict__ Wr1,
    const float* __restrict__ Wr2, _Float16* __restrict__ wth)
{
    __shared__ float tile[16][17];   // pad to kill bank conflicts
    const int p = blockIdx.z;
    const float* W = (p == 0) ? Wu : (p == 1) ? Wr1 : Wr2;
    _Float16* wt = wth + (size_t)p * H_ * D_;

    const int n0 = blockIdx.x * 16;
    const int k0 = blockIdx.y * 16;
    const int tx = threadIdx.x & 15;
    const int ty = threadIdx.x >> 4;

    tile[ty][tx] = W[(size_t)(k0 + ty) * H_ + (n0 + tx)];            // coalesced read
    __syncthreads();
    wt[(size_t)(n0 + ty) * D_ + (k0 + tx)] = (_Float16)tile[tx][ty]; // coalesced write
}

// ---------------------------------------------------------------------------
// Kernel 1: fused projection GEMM, LDS double-buffered B panel + register
// software pipeline for both A and B fragments.
//   out[p] = act( xh[16384,512] * wt[p]^T + bias[p] ),  act = tanh for p==0.
//
// Block = 256 threads = 8 waves; block tile = 128 rows x 64 cols; each wave
// owns 16 rows x 64 cols (4 f32 accumulators). Per chunk of K=128:
//   [issue global B loads for chunk+1] -> [4 K-steps; each prefetches the
//   NEXT step's A (global) and B (LDS) fragments into registers before its
//   4 WMMAs] -> [ds_store chunk+1] -> barrier.
// A-fragment prefetch crosses chunk boundaries (it's pure global traffic),
// so step 0 of a chunk never stalls on LOADcnt behind the big pre-B fetch.
// ---------------------------------------------------------------------------
__global__ __launch_bounds__(256) void fused_proj_gemm_kernel(
    const _Float16* __restrict__ xh,
    const _Float16* __restrict__ wth,
    const float* __restrict__ bu, const float* __restrict__ br1,
    const float* __restrict__ br2,
    float* __restrict__ ws)
{
    __shared__ _Float16 bsm[2][64][BSTRIDE];     // 2 x 64 x 136 f16 = 34,816 B

    const int tid  = threadIdx.x;
    const int wave = tid >> 5;
    const int lane = tid & 31;
    const int p    = blockIdx.z;

    const float*     bias = (p == 0) ? bu : (p == 1) ? br1 : br2;
    const _Float16*  wt   = wth + (size_t)p * H_ * D_;
    float*           out  = ws + (size_t)p * PLANE;

    const int m0  = blockIdx.x * 128 + wave * 16;
    const int n0  = blockIdx.y * 64;
    const int r16 = lane & 15;
    const int hi  = lane >> 4;          // which 16-lane half of the wave

    // Cooperative-fill assignment: 4 threads per column, each moves 4 x v8h.
    const int fcol = tid >> 2;                    // 0..63
    const int fsub = tid & 3;                     // 0..3
    const _Float16* fsrc = wt + (size_t)(n0 + fcol) * D_ + fsub * 8;

    v8f acc[4] = {v8f{}, v8f{}, v8f{}, v8f{}};
    const _Float16* arow = xh + (size_t)(m0 + r16) * D_;

    // --- prologue: A fragment for K-step 0, and fill stage 0 (K=[0,128)) ---
    v16h aCur = load_frag16(arow + hi * 16, arow + hi * 16 + 8);
    {
        v8h pre[4];
        #pragma unroll
        for (int j = 0; j < 4; ++j)
            pre[j] = *(const v8h*)(fsrc + j * 32);
        #pragma unroll
        for (int j = 0; j < 4; ++j)
            *(v8h*)&bsm[0][fcol][fsub * 8 + j * 32] = pre[j];
    }
    __syncthreads();

    int stage = 0;
    #pragma unroll
    for (int chunk = 0; chunk < NCHUNK; ++chunk) {
        const int  kbase = chunk * CHUNK_K;
        const bool more  = (chunk + 1) < NCHUNK;

        // Issue next chunk's global B loads up front (hidden by 16 WMMAs).
        v8h pre[4];
        if (more) {
            const _Float16* nsrc = fsrc + (kbase + CHUNK_K);
            #pragma unroll
            for (int j = 0; j < 4; ++j)
                pre[j] = *(const v8h*)(nsrc + j * 32);
        }

        // B fragments for step 0 of this chunk (LDS just became valid).
        const _Float16* bbase = &bsm[stage][r16][hi * 16];
        v16h bCur[4];
        #pragma unroll
        for (int t = 0; t < 4; ++t)
            bCur[t] = load_frag16(bbase + t * 16 * BSTRIDE,
                                  bbase + t * 16 * BSTRIDE + 8);

        #pragma unroll
        for (int s = 0; s < CHUNK_K / 32; ++s) {
            const int ks = chunk * (CHUNK_K / 32) + s;   // global step 0..15

            // ---- prefetch next step's fragments into registers ----
            v16h aNxt;
            v16h bNxt[4];
            const bool moreA = (ks + 1) < NSTEPS;        // A crosses chunks
            const bool moreB = (s + 1) < (CHUNK_K / 32); // B stays in chunk
            if (moreA) {
                const int ka = (ks + 1) * 32 + hi * 16;
                aNxt = load_frag16(arow + ka, arow + ka + 8);
            }
            if (moreB) {
                const _Float16* bn = bbase + (s + 1) * 32;
                #pragma unroll
                for (int t = 0; t < 4; ++t)
                    bNxt[t] = load_frag16(bn + t * 16 * BSTRIDE,
                                          bn + t * 16 * BSTRIDE + 8);
            }

            // ---- 4 WMMAs on current fragments ----
            #pragma unroll
            for (int t = 0; t < 4; ++t)
                acc[t] = __builtin_amdgcn_wmma_f32_16x16x32_f16(
                    false, aCur, false, bCur[t], (short)0, acc[t], false, false);

            // ---- rotate ----
            if (moreA) aCur = aNxt;
            if (moreB) {
                #pragma unroll
                for (int t = 0; t < 4; ++t) bCur[t] = bNxt[t];
            }
        }

        // Store the prefetched chunk into the other stage, then barrier.
        if (more) {
            #pragma unroll
            for (int j = 0; j < 4; ++j)
                *(v8h*)&bsm[stage ^ 1][fcol][fsub * 8 + j * 32] = pre[j];
        }
        __syncthreads();
        stage ^= 1;
    }

    // Epilogue: bias (+ tanh for the u projection). C/D layout: element r of
    // the v8f maps to row M = r + 8*hi, column N = lane&15.
    #pragma unroll
    for (int t = 0; t < 4; ++t) {
        const int   ncol = n0 + t * 16 + r16;
        const float bv   = bias[ncol];
        #pragma unroll
        for (int r = 0; r < 8; ++r) {
            float v = acc[t][r] + bv;
            if (p == 0) v = tanhf(v);
            out[(size_t)(m0 + hi * 8 + r) * H_ + ncol] = v;
        }
    }
}

// ---------------------------------------------------------------------------
// Kernel 2: row softmax over H for planes 1 (r1) and 2 (r2), in place.
// One block (256 threads) per row of 512; 2 elements per thread.
// ---------------------------------------------------------------------------
__global__ __launch_bounds__(256) void softmax_rows_kernel(float* __restrict__ ws)
{
    float* row = ws + (size_t)(1 + blockIdx.y) * PLANE + (size_t)blockIdx.x * H_;
    const int tid = threadIdx.x;

    float v0 = row[tid];
    float v1 = row[tid + 256];

    __shared__ float red[8];

    // --- max reduction ---
    float m = fmaxf(v0, v1);
    #pragma unroll
    for (int off = 16; off > 0; off >>= 1)
        m = fmaxf(m, __shfl_xor(m, off, 32));
    if ((tid & 31) == 0) red[tid >> 5] = m;
    __syncthreads();
    m = red[0];
    #pragma unroll
    for (int i = 1; i < 8; ++i) m = fmaxf(m, red[i]);
    __syncthreads();

    // --- exp + sum reduction ---
    const float e0 = __expf(v0 - m);
    const float e1 = __expf(v1 - m);
    float s = e0 + e1;
    #pragma unroll
    for (int off = 16; off > 0; off >>= 1)
        s += __shfl_xor(s, off, 32);
    if ((tid & 31) == 0) red[tid >> 5] = s;
    __syncthreads();
    s = red[0];
    #pragma unroll
    for (int i = 1; i < 8; ++i) s += red[i];

    const float inv = 1.0f / s;
    row[tid]       = e0 * inv;
    row[tid + 256] = e1 * inv;
}

// ---------------------------------------------------------------------------
// Kernel 3: sequential gated scan over T. One thread per (b, h) — 16384
// independent recurrences, loads/stores coalesced across h. The three loads
// per step are independent of the carried state, so they pipeline deeply.
// ---------------------------------------------------------------------------
__global__ __launch_bounds__(256) void scan_kernel(
    const float* __restrict__ ws, float* __restrict__ out)
{
    const int idx = blockIdx.x * 256 + threadIdx.x;   // 0 .. B_*H_-1
    const int b = idx >> 9;                           // / 512
    const int h = idx & (H_ - 1);

    const size_t base = (size_t)b * T_ * H_ + h;
    const float* u  = ws + base;                      // plane 0 (tanh already applied)
    const float* r1 = ws + PLANE + base;              // plane 1 (softmaxed)
    const float* r2 = ws + 2 * PLANE + base;          // plane 2 (softmaxed)
    float* o = out + base;

    float s = 0.0f;
    #pragma unroll 4
    for (int t = 0; t < T_; ++t) {
        const size_t off = (size_t)t * H_;
        s = fmaf(r1[off], s, r2[off] * u[off]);
        o[off] = s;
    }
}

// ---------------------------------------------------------------------------
// Launch
// ---------------------------------------------------------------------------
extern "C" void kernel_launch(void* const* d_in, const int* in_sizes, int n_in,
                              void* d_out, int out_size, void* d_ws, size_t ws_size,
                              hipStream_t stream) {
    const float* x   = (const float*)d_in[0];
    const float* Wu  = (const float*)d_in[1];
    const float* Wr1 = (const float*)d_in[2];
    const float* Wr2 = (const float*)d_in[3];
    const float* bu  = (const float*)d_in[4];
    const float* br1 = (const float*)d_in[5];
    const float* br2 = (const float*)d_in[6];
    float* out = (float*)d_out;

    // Workspace layout:
    //   [0, 3*PLANE) f32             : u / r1 / r2 planes
    //   then PLANE f16               : x converted to f16
    //   then 3 * H_*D_ f16           : transposed f16 weights
    char* wsb = (char*)d_ws;
    float*     wsf = (float*)wsb;
    _Float16*  xh  = (_Float16*)(wsb + 3 * PLANE * sizeof(float));
    _Float16*  wth = (_Float16*)(wsb + 3 * PLANE * sizeof(float) + PLANE * sizeof(_Float16));

    // 0a: x -> f16 (8,388,608 elems, 4 per thread)
    convert_x_kernel<<<dim3((M_TOT * D_) / (256 * 4)), dim3(256), 0, stream>>>(x, xh);

    // 0b: W -> W^T in f16 (3 matrices, 16x16 tiles)
    transpose_w_kernel<<<dim3(H_ / 16, D_ / 16, 3), dim3(256), 0, stream>>>(Wu, Wr1, Wr2, wth);

    // 1: WMMA GEMM + bias (+ tanh for u), LDS double-buffered + reg pipelined
    fused_proj_gemm_kernel<<<dim3(M_TOT / 128, H_ / 64, 3), dim3(256), 0, stream>>>(
        xh, wth, bu, br1, br2, wsf);

    // 2: softmax rows for r1 and r2
    softmax_rows_kernel<<<dim3(M_TOT, 2), dim3(256), 0, stream>>>(wsf);

    // 3: sequential scan -> output
    scan_kernel<<<dim3((B_ * H_) / 256), dim3(256), 0, stream>>>(wsf, out);
}
